// KipfAndWillingConv_35081292874224
// MI455X (gfx1250) — matrix operand order repaired and checked by reading, not compiled, not run on hardware
//
#include <hip/hip_runtime.h>

#define N_FEAT 512
#define N_FILT 512

typedef __attribute__((ext_vector_type(8)))  __bf16 v8bf;
typedef __attribute__((ext_vector_type(16))) __bf16 v16bf;
typedef __attribute__((ext_vector_type(8)))  float  v8f;

// ---------------------------------------------------------------------------
// Pack helpers: fp32 -> (bf16 hi, bf16 lo residual), two at a time into dwords
// ---------------------------------------------------------------------------
static __device__ __forceinline__ void cvt_pack8(const float* v,
                                                 unsigned int* hi,
                                                 unsigned int* lo)
{
#pragma unroll
    for (int i = 0; i < 4; ++i) {
        float a = v[2 * i], b = v[2 * i + 1];
        __bf16 ha = (__bf16)a, hb = (__bf16)b;
        __bf16 la = (__bf16)(a - (float)ha);
        __bf16 lb = (__bf16)(b - (float)hb);
        hi[i] = (unsigned int)__builtin_bit_cast(unsigned short, ha) |
                ((unsigned int)__builtin_bit_cast(unsigned short, hb) << 16);
        lo[i] = (unsigned int)__builtin_bit_cast(unsigned short, la) |
                ((unsigned int)__builtin_bit_cast(unsigned short, lb) << 16);
    }
}

// ---------------------------------------------------------------------------
// W transpose (512x512, 1 MB, one-time): Wt[n][k] = W[k][n]
// ---------------------------------------------------------------------------
__global__ __launch_bounds__(256)
void transpose_w(const float* __restrict__ W, float* __restrict__ Wt)
{
    __shared__ float tile[32][33];
    const int tx = threadIdx.x;          // 0..31
    const int ty = threadIdx.y;          // 0..7
    const int x = blockIdx.x * 32 + tx;
    const int y = blockIdx.y * 32;
#pragma unroll
    for (int j = 0; j < 4; ++j)
        tile[ty + j * 8][tx] = W[(size_t)(y + ty + j * 8) * N_FILT + x];
    __syncthreads();
    const int ox = blockIdx.y * 32 + tx;
    const int oy = blockIdx.x * 32;
#pragma unroll
    for (int j = 0; j < 4; ++j)
        Wt[(size_t)(oy + ty + j * 8) * N_FEAT + ox] = tile[tx][ty + j * 8];
}

// ---------------------------------------------------------------------------
// GEMM: XF = X[M x 512] * W[512 x 512] with fp32 emulated via bf16 hi/lo
// split (hi*hi + hi*lo + lo*hi), f32 WMMA accumulation.
// Block tile 128(M) x 64(N), K-step 32. 256 threads = 8 waves (4 M x 2 N),
// each wave computes a 32x32 output (4 accumulators, 12 WMMAs / K-step).
// B is read from pre-transposed Wt so BOTH tiles stage as packed b128.
// ---------------------------------------------------------------------------
#define BLK_M 128
#define BLK_N 64
#define BLK_K 32
#define LDS_LD 40   // 32 + 8 pad; row stride 80 B keeps 16 B alignment

__global__ __launch_bounds__(256)
void gcn_gemm_xw(const float* __restrict__ X, const float* __restrict__ Wt,
                 float* __restrict__ XF, int M)
{
    __shared__ alignas(16) __bf16 Ah[BLK_M][LDS_LD];
    __shared__ alignas(16) __bf16 Al[BLK_M][LDS_LD];
    __shared__ alignas(16) __bf16 Bh[BLK_N][LDS_LD];   // [n][k]
    __shared__ alignas(16) __bf16 Bl[BLK_N][LDS_LD];

    const int tid  = threadIdx.x;
    const int lane = tid & 31;
    const int wave = tid >> 5;      // 0..7
    const int wm   = wave & 3;      // -> M base wm*32 (subtiles +0, +16)
    const int wn   = wave >> 2;     // -> N base wn*32 (subtiles +0, +16)

    const int block_m = blockIdx.y * BLK_M;
    const int block_n = blockIdx.x * BLK_N;

    // A staging: 128x32 fp32 = 16 per thread (one 16-float run per row half)
    const int arow = tid >> 1;              // 0..127
    const int acol = (tid & 1) * 16;        // 0 or 16
    // B staging: 64x32 fp32 (from Wt) = 8 per thread
    const int brow = tid >> 2;              // 0..63   (n)
    const int bcol = (tid & 3) * 8;         // 0..24   (k)

    const int lr = lane & 15;               // row (A) / col (B,C) in tile
    const int kg = lane >> 4;               // K-half selector

    v8f acc00 = {}, acc01 = {};             // m0 x {n0, n1}
    v8f acc10 = {}, acc11 = {};             // m1 x {n0, n1}

    for (int k0 = 0; k0 < N_FEAT; k0 += BLK_K) {
        // ---- stage A tile 128x32 (rows guarded vs M), packed b128 stores --
        {
            const int gr = block_m + arow;
            float v[16];
            if (gr < M) {
                const float* p = X + (size_t)gr * N_FEAT + k0 + acol;
                float4 f0 = *(const float4*)(p + 0);
                float4 f1 = *(const float4*)(p + 4);
                float4 f2 = *(const float4*)(p + 8);
                float4 f3 = *(const float4*)(p + 12);
                v[0]=f0.x; v[1]=f0.y; v[2]=f0.z; v[3]=f0.w;
                v[4]=f1.x; v[5]=f1.y; v[6]=f1.z; v[7]=f1.w;
                v[8]=f2.x; v[9]=f2.y; v[10]=f2.z; v[11]=f2.w;
                v[12]=f3.x; v[13]=f3.y; v[14]=f3.z; v[15]=f3.w;
            } else {
#pragma unroll
                for (int i = 0; i < 16; ++i) v[i] = 0.f;
            }
            unsigned int hi[8], lo[8];
            cvt_pack8(v,     hi,     lo);
            cvt_pack8(v + 8, hi + 4, lo + 4);
            *(uint4*)&Ah[arow][acol]     = *(const uint4*)(hi);
            *(uint4*)&Ah[arow][acol + 8] = *(const uint4*)(hi + 4);
            *(uint4*)&Al[arow][acol]     = *(const uint4*)(lo);
            *(uint4*)&Al[arow][acol + 8] = *(const uint4*)(lo + 4);
        }
        // ---- stage B tile 64x32 from Wt[n][k], packed b128 stores ---------
        {
            const float* p = Wt + (size_t)(block_n + brow) * N_FEAT + k0 + bcol;
            float4 f0 = *(const float4*)(p + 0);
            float4 f1 = *(const float4*)(p + 4);
            float v[8] = {f0.x, f0.y, f0.z, f0.w, f1.x, f1.y, f1.z, f1.w};
            unsigned int hi[4], lo[4];
            cvt_pack8(v, hi, lo);
            *(uint4*)&Bh[brow][bcol] = *(const uint4*)(hi);
            *(uint4*)&Bl[brow][bcol] = *(const uint4*)(lo);
        }
        __syncthreads();

        // ---- A fragments for both M subtiles (16x32 bf16 wave32 layout:
        //      lane<16 -> K 0..7 & 16..23, lane>=16 -> K 8..15 & 24..31) ----
        const int ar0 = wm * 32 + lr;
        const int ar1 = ar0 + 16;
        v16bf a0h = __builtin_shufflevector(
            *(const v8bf*)&Ah[ar0][kg * 8], *(const v8bf*)&Ah[ar0][16 + kg * 8],
            0,1,2,3,4,5,6,7,8,9,10,11,12,13,14,15);
        v16bf a0l = __builtin_shufflevector(
            *(const v8bf*)&Al[ar0][kg * 8], *(const v8bf*)&Al[ar0][16 + kg * 8],
            0,1,2,3,4,5,6,7,8,9,10,11,12,13,14,15);
        v16bf a1h = __builtin_shufflevector(
            *(const v8bf*)&Ah[ar1][kg * 8], *(const v8bf*)&Ah[ar1][16 + kg * 8],
            0,1,2,3,4,5,6,7,8,9,10,11,12,13,14,15);
        v16bf a1l = __builtin_shufflevector(
            *(const v8bf*)&Al[ar1][kg * 8], *(const v8bf*)&Al[ar1][16 + kg * 8],
            0,1,2,3,4,5,6,7,8,9,10,11,12,13,14,15);

        // ---- B fragments for both N subtiles (32x16 layout: lane<16 ->
        //      K 0..7 & 16..23, lane>=16 -> K 8..15 & 24..31; col = lane%16) -
        const int n0 = wn * 32 + lr;
        const int n1 = n0 + 16;
        v16bf b0h = __builtin_shufflevector(
            *(const v8bf*)&Bh[n0][kg * 8], *(const v8bf*)&Bh[n0][16 + kg * 8],
            0,1,2,3,4,5,6,7,8,9,10,11,12,13,14,15);
        v16bf b0l = __builtin_shufflevector(
            *(const v8bf*)&Bl[n0][kg * 8], *(const v8bf*)&Bl[n0][16 + kg * 8],
            0,1,2,3,4,5,6,7,8,9,10,11,12,13,14,15);
        v16bf b1h = __builtin_shufflevector(
            *(const v8bf*)&Bh[n1][kg * 8], *(const v8bf*)&Bh[n1][16 + kg * 8],
            0,1,2,3,4,5,6,7,8,9,10,11,12,13,14,15);
        v16bf b1l = __builtin_shufflevector(
            *(const v8bf*)&Bl[n1][kg * 8], *(const v8bf*)&Bl[n1][16 + kg * 8],
            0,1,2,3,4,5,6,7,8,9,10,11,12,13,14,15);

        // ---- 12 WMMAs: 3-product fp32 emulation x 4 output subtiles ------
        acc00 = __builtin_amdgcn_wmma_f32_16x16x32_bf16(false, a0h, false, b0h, (short)0, acc00, false, false);
        acc00 = __builtin_amdgcn_wmma_f32_16x16x32_bf16(false, a0h, false, b0l, (short)0, acc00, false, false);
        acc00 = __builtin_amdgcn_wmma_f32_16x16x32_bf16(false, a0l, false, b0h, (short)0, acc00, false, false);

        acc01 = __builtin_amdgcn_wmma_f32_16x16x32_bf16(false, a0h, false, b1h, (short)0, acc01, false, false);
        acc01 = __builtin_amdgcn_wmma_f32_16x16x32_bf16(false, a0h, false, b1l, (short)0, acc01, false, false);
        acc01 = __builtin_amdgcn_wmma_f32_16x16x32_bf16(false, a0l, false, b1h, (short)0, acc01, false, false);

        acc10 = __builtin_amdgcn_wmma_f32_16x16x32_bf16(false, a1h, false, b0h, (short)0, acc10, false, false);
        acc10 = __builtin_amdgcn_wmma_f32_16x16x32_bf16(false, a1h, false, b0l, (short)0, acc10, false, false);
        acc10 = __builtin_amdgcn_wmma_f32_16x16x32_bf16(false, a1l, false, b0h, (short)0, acc10, false, false);

        acc11 = __builtin_amdgcn_wmma_f32_16x16x32_bf16(false, a1h, false, b1h, (short)0, acc11, false, false);
        acc11 = __builtin_amdgcn_wmma_f32_16x16x32_bf16(false, a1h, false, b1l, (short)0, acc11, false, false);
        acc11 = __builtin_amdgcn_wmma_f32_16x16x32_bf16(false, a1l, false, b1h, (short)0, acc11, false, false);

        __syncthreads();
    }

    // ---- store: VGPR g -> M = g + (lane>=16 ? 8 : 0), N = lane%16 --------
    const int cbase = block_n + wn * 32 + lr;
#pragma unroll
    for (int g = 0; g < 8; ++g) {
        const int r0 = block_m + wm * 32 + kg * 8 + g;
        const int r1 = r0 + 16;
        if (r0 < M) {
            float* op = XF + (size_t)r0 * N_FILT + cbase;
            op[0]  = acc00[g];
            op[16] = acc01[g];
        }
        if (r1 < M) {
            float* op = XF + (size_t)r1 * N_FILT + cbase;
            op[0]  = acc10[g];
            op[16] = acc11[g];
        }
    }
}

// ---------------------------------------------------------------------------
// Zero the output (harness poisons d_out before timing).
// ---------------------------------------------------------------------------
__global__ __launch_bounds__(256)
void zero_out(float4* __restrict__ out, int n4)
{
    int i = blockIdx.x * 256 + threadIdx.x;
    if (i < n4) out[i] = make_float4(0.f, 0.f, 0.f, 0.f);
}

// ---------------------------------------------------------------------------
// Edge scatter: out[dst] += w * XF[src].  One wave (32 lanes) per edge,
// float4 gathers + per-lane non-returning global_atomic_add_f32.
// ---------------------------------------------------------------------------
__global__ __launch_bounds__(256)
void gcn_scatter(const float* __restrict__ XF,
                 const int* __restrict__ esrc, const int* __restrict__ edst,
                 const float* __restrict__ ew, float* __restrict__ out, int nE)
{
    const int wave = threadIdx.x >> 5;
    const int lane = threadIdx.x & 31;
    const int e = blockIdx.x * 8 + wave;
    if (e >= nE) return;

    const int   s  = esrc[e];
    const int   d  = edst[e];
    const float we = ew[e];

    const float4* xp = (const float4*)(XF + (size_t)s * N_FILT);
    float*        op = out + (size_t)d * N_FILT;

#pragma unroll
    for (int it = 0; it < N_FILT / 128; ++it) {   // 4 iterations
        const int f4 = it * 32 + lane;            // float4 index 0..127
        const float4 v = xp[f4];
        const int f = f4 * 4;
        atomicAdd(op + f + 0, v.x * we);
        atomicAdd(op + f + 1, v.y * we);
        atomicAdd(op + f + 2, v.z * we);
        atomicAdd(op + f + 3, v.w * we);
    }
}

// ---------------------------------------------------------------------------
extern "C" void kernel_launch(void* const* d_in, const int* in_sizes, int n_in,
                              void* d_out, int out_size, void* d_ws, size_t ws_size,
                              hipStream_t stream)
{
    const float* x   = (const float*)d_in[0];   // [M, 512]
    const float* w   = (const float*)d_in[1];   // [512, 512]
    const int*   es  = (const int*)  d_in[2];   // [E]
    const int*   ed  = (const int*)  d_in[3];   // [E]
    const float* ewt = (const float*)d_in[4];   // [E]
    float*       out = (float*)d_out;           // [M, 512]

    const int M  = in_sizes[0] / N_FEAT;
    const int nE = in_sizes[2];

    // workspace layout: XF [M*512] floats, then Wt [512*512] floats
    float* XF = (float*)d_ws;
    float* Wt = XF + (size_t)M * N_FILT;

    // 1) zero output
    {
        const int n4 = out_size / 4;
        zero_out<<<(n4 + 255) / 256, 256, 0, stream>>>((float4*)out, n4);
    }
    // 2) Wt = W^T (one-time, 1 MB; makes GEMM B-tile K-contiguous)
    {
        dim3 grid(N_FILT / 32, N_FEAT / 32), blk(32, 8);
        transpose_w<<<grid, blk, 0, stream>>>(w, Wt);
    }
    // 3) XF = X @ W  (bf16-split WMMA, f32 accumulate)
    {
        dim3 grid(N_FILT / BLK_N, (M + BLK_M - 1) / BLK_M);
        gcn_gemm_xw<<<grid, 256, 0, stream>>>(x, Wt, XF, M);
    }
    // 4) out[dst] += w * XF[src]
    {
        gcn_scatter<<<(nE + 7) / 8, 256, 0, stream>>>(XF, es, ed, ewt, out, nE);
    }
}
